// VectorQuantizer_39221641347223
// MI455X (gfx1250) — matrix operand order repaired and checked by reading, not compiled, not run on hardware
//
#include <hip/hip_runtime.h>
#include <hip/hip_bf16.h>

// ---------------------------------------------------------------------------
// VQ (rotation trick) for MI455X / gfx1250.
// x: [8,256,64,64] f32, codebook e_i_ts: [256,8192] f32.
// Distance GEMM (32768x8192x256, 137 GFLOP) fused with argmin using
// v_wmma_f32_16x16x32_bf16. Codebook tiles double-buffered in LDS via
// gfx1250 async copies (global_load_async_to_lds_b128 / ASYNCcnt); each
// wave owns 32 rows so every LDS B-fragment feeds two WMMAs (LDS-port
// balanced). Rotation trick reduced to out = alpha*e + beta*q.
// ---------------------------------------------------------------------------

typedef __bf16 bf16_t;
typedef __attribute__((ext_vector_type(16))) __bf16 v16bf;
typedef __attribute__((ext_vector_type(8)))  __bf16 v8bf;
typedef __attribute__((ext_vector_type(8)))  float  v8f;

#define D_DIM   256
#define K_CODES 8192
#define HW      4096
#define B_DIM   8
#define N_ROWS  32768          // B*H*W
#define EPSF    1e-6f

// LDS tile: 64 codes x 256 d (bf16), row padded 512B -> 528B (bank-conflict free)
#define ROW_BF16 264           // 264 bf16 = 528 bytes = 132 dwords (4 banks skew)
#define TILE_BF16 (64 * ROW_BF16)

// 16-byte async copy global -> LDS (fire and forget, ASYNCcnt)
__device__ __forceinline__ void async_copy_b128(unsigned lds_off, const void* gptr) {
    asm volatile("global_load_async_to_lds_b128 %0, %1, off"
                 :: "v"(lds_off), "v"(gptr) : "memory");
}

// ---------------------------------------------------------------------------
// Kernel 1: transpose x [B, D, HW] -> A_f32 [N, D] and A_bf16 [N, D]
// ---------------------------------------------------------------------------
__global__ __launch_bounds__(256)
void vq_transpose_x(const float* __restrict__ x,
                    float* __restrict__ Af, bf16_t* __restrict__ Ab) {
    __shared__ float tile[32][33];
    const int b   = blockIdx.z;
    const int hw0 = blockIdx.x * 32;
    const int d0  = blockIdx.y * 32;
    const int tx  = threadIdx.x;   // 0..31
    const int ty  = threadIdx.y;   // 0..7
    const float* src = x + ((size_t)b * D_DIM + d0) * HW + hw0;
#pragma unroll
    for (int r = 0; r < 4; ++r)
        tile[ty + r * 8][tx] = src[(size_t)(ty + r * 8) * HW + tx];
    __syncthreads();
#pragma unroll
    for (int r = 0; r < 4; ++r) {
        const int hwl = ty + r * 8;
        const float v = tile[tx][hwl];
        const size_t o = ((size_t)b * HW + hw0 + hwl) * D_DIM + d0 + tx;
        Af[o] = v;
        Ab[o] = (bf16_t)v;
    }
}

// ---------------------------------------------------------------------------
// Kernel 2: transpose codebook [D, K] -> B_bf16 [K, D]
// ---------------------------------------------------------------------------
__global__ __launch_bounds__(256)
void vq_transpose_cb(const float* __restrict__ e, bf16_t* __restrict__ Bb) {
    __shared__ float tile[32][33];
    const int k0 = blockIdx.x * 32;
    const int d0 = blockIdx.y * 32;
    const int tx = threadIdx.x;
    const int ty = threadIdx.y;
    const float* src = e + (size_t)d0 * K_CODES + k0;
#pragma unroll
    for (int r = 0; r < 4; ++r)
        tile[ty + r * 8][tx] = src[(size_t)(ty + r * 8) * K_CODES + tx];
    __syncthreads();
#pragma unroll
    for (int r = 0; r < 4; ++r) {
        const int kl = ty + r * 8;
        Bb[(size_t)(k0 + kl) * D_DIM + d0 + tx] = (bf16_t)tile[tx][kl];
    }
}

// ---------------------------------------------------------------------------
// Kernel 3: codebook column squared norms (f32), coalesced over k
// ---------------------------------------------------------------------------
__global__ __launch_bounds__(256)
void vq_cnorm(const float* __restrict__ e, float* __restrict__ cn) {
    const int k = blockIdx.x * 256 + threadIdx.x;
    float s = 0.f;
#pragma unroll 4
    for (int d = 0; d < D_DIM; ++d) {
        const float v = e[(size_t)d * K_CODES + k];
        s += v * v;
    }
    cn[k] = s;
}

// ---------------------------------------------------------------------------
// Kernel 4: fused bf16 WMMA GEMM + argmin.
// Block = 256 threads = 8 waves; wave owns 32 rows (2 A-tiles) -> block 256
// rows. 64-code tiles double-buffered in LDS; each B fragment feeds 2 WMMAs.
// dist = cnorm[k] - 2*(x . e_k)   (||x||^2 is row-constant, dropped)
// ---------------------------------------------------------------------------
__global__ __launch_bounds__(256)
void vq_argmin_wmma(const bf16_t* __restrict__ A, const bf16_t* __restrict__ Bm,
                    const float* __restrict__ cnorm, int* __restrict__ idx_out) {
    __shared__ bf16_t btile[2][TILE_BF16];    // 2 x 33792 B

    const int lane = threadIdx.x & 31;
    const int wave = threadIdx.x >> 5;
    const int n0   = blockIdx.x * 256 + wave * 32;

    // ---- staging geometry: thread copies 8 x 16B chunks per tile ----
    const int scol = threadIdx.x & 31;        // 16B column within 512B row
    const int srow = threadIdx.x >> 5;        // starting row (stride 8)
    const unsigned lbuf0 = (unsigned)(uintptr_t)(&btile[0][0]);
    const unsigned lbuf1 = (unsigned)(uintptr_t)(&btile[1][0]);

    // ---- resident A fragments: two 16-row tiles, 8 d-steps each ----
    const int dlo = (lane & 16) ? 8 : 0;
    v16bf afrag0[8], afrag1[8];
#pragma unroll
    for (int ds = 0; ds < 8; ++ds) {
        const bf16_t* rp0 = A + (size_t)(n0 + (lane & 15)) * D_DIM + ds * 32 + dlo;
        const bf16_t* rp1 = rp0 + 16 * D_DIM;
        const v8bf p0 = *(const v8bf*)(rp0);
        const v8bf p1 = *(const v8bf*)(rp0 + 16);
        const v8bf p2 = *(const v8bf*)(rp1);
        const v8bf p3 = *(const v8bf*)(rp1 + 16);
        v16bf a0, a1;
#pragma unroll
        for (int i = 0; i < 8; ++i) {
            a0[i] = p0[i]; a0[i + 8] = p1[i];
            a1[i] = p2[i]; a1[i + 8] = p3[i];
        }
        afrag0[ds] = a0;
        afrag1[ds] = a1;
    }

    float best0[8], best1[8];
    int   bidx0[8], bidx1[8];
#pragma unroll
    for (int v = 0; v < 8; ++v) {
        best0[v] = 3.4e38f; bidx0[v] = 0;
        best1[v] = 3.4e38f; bidx1[v] = 0;
    }

    // B fragment lane mapping: col = lane&15, d-base = (lane&16 ? 8 : 0)
    const int bcol = lane & 15;
    const int bd   = (lane & 16) ? 8 : 0;

    // ---- prologue: stage tile 0 into buffer 0 ----
#pragma unroll
    for (int i = 0; i < 8; ++i) {
        const int row = srow + i * 8;
        async_copy_b128(lbuf0 + row * 528 + scol * 16,
                        Bm + ((size_t)row << 8) + scol * 8);
    }

    const int NT = K_CODES / 64;              // 128 tiles
    for (int it = 0; it < NT; ++it) {
        const int cur = it & 1;
        // stage next tile into the other buffer (prev readers passed barrier)
        if (it + 1 < NT) {
            const unsigned lb = cur ? lbuf0 : lbuf1;
            const int ktn = (it + 1) * 64;
#pragma unroll
            for (int i = 0; i < 8; ++i) {
                const int row = srow + i * 8;
                async_copy_b128(lb + row * 528 + scol * 16,
                                Bm + ((size_t)(ktn + row) << 8) + scol * 8);
            }
            asm volatile("s_wait_asynccnt 0x8" ::: "memory"); // oldest 8 done
        } else {
            asm volatile("s_wait_asynccnt 0x0" ::: "memory");
        }
        __syncthreads();                       // current tile visible to block

        const bf16_t* bt = &btile[cur][0];
        const int kt = it * 64;

        v8f acc0[4], acc1[4];
        const v8f zero = {};
#pragma unroll
        for (int c = 0; c < 4; ++c) { acc0[c] = zero; acc1[c] = zero; }

#pragma unroll
        for (int ds = 0; ds < 8; ++ds) {
#pragma unroll
            for (int c = 0; c < 4; ++c) {
                const bf16_t* bp = bt + (c * 16 + bcol) * ROW_BF16 + ds * 32 + bd;
                const v8bf q0 = *(const v8bf*)(bp);
                const v8bf q1 = *(const v8bf*)(bp + 16);
                v16bf bfr;
#pragma unroll
                for (int i = 0; i < 8; ++i) { bfr[i] = q0[i]; bfr[i + 8] = q1[i]; }
                acc0[c] = __builtin_amdgcn_wmma_f32_16x16x32_bf16(
                    false, afrag0[ds], false, bfr, (short)0, acc0[c], false, false);
                acc1[c] = __builtin_amdgcn_wmma_f32_16x16x32_bf16(
                    false, afrag1[ds], false, bfr, (short)0, acc1[c], false, false);
            }
        }

        // argmin update; C layout: VGPR v -> row v (lanes 0-15) or v+8,
        // column = lane&15 within the 16-code subtile
#pragma unroll
        for (int c = 0; c < 4; ++c) {
            const int code = kt + c * 16 + bcol;
            const float cn = cnorm[code];
#pragma unroll
            for (int v = 0; v < 8; ++v) {
                const float d0 = cn - 2.0f * acc0[c][v];
                if (d0 < best0[v]) { best0[v] = d0; bidx0[v] = code; }
                const float d1 = cn - 2.0f * acc1[c][v];
                if (d1 < best1[v]) { best1[v] = d1; bidx1[v] = code; }
            }
        }
        __syncthreads();                       // done reading buf before reuse
    }

    // ---- reduce across the 16 lanes holding each row; write both tiles ----
#pragma unroll
    for (int v = 0; v < 8; ++v) {
        float bv0 = best0[v], bv1 = best1[v];
        int   bi0 = bidx0[v], bi1 = bidx1[v];
#pragma unroll
        for (int m = 1; m < 16; m <<= 1) {
            const float ov0 = __shfl_xor(bv0, m, 32);
            const int   oi0 = __shfl_xor(bi0, m, 32);
            if (ov0 < bv0 || (ov0 == bv0 && oi0 < bi0)) { bv0 = ov0; bi0 = oi0; }
            const float ov1 = __shfl_xor(bv1, m, 32);
            const int   oi1 = __shfl_xor(bi1, m, 32);
            if (ov1 < bv1 || (ov1 == bv1 && oi1 < bi1)) { bv1 = ov1; bi1 = oi1; }
        }
        if ((lane & 15) == 0) {
            const int half = (lane >> 4) << 3;     // 0 or 8
            idx_out[n0 + half + v]      = bi0;
            idx_out[n0 + 16 + half + v] = bi1;
        }
    }
}

// ---------------------------------------------------------------------------
// Kernel 5: per-row rotation-trick scalars (alpha, beta).
// ---------------------------------------------------------------------------
__global__ __launch_bounds__(256)
void vq_rowstats(const float* __restrict__ Af, const float* __restrict__ e,
                 const int* __restrict__ idx, float* __restrict__ ab) {
    const int lane = threadIdx.x & 31;
    const int n    = blockIdx.x * 8 + (threadIdx.x >> 5);
    const int k    = idx[n];
    const float* er = Af + (size_t)n * D_DIM;
    float ee = 0.f, qq = 0.f, eq = 0.f;
#pragma unroll
    for (int j = 0; j < 8; ++j) {
        const int d = j * 32 + lane;
        const float ev = er[d];
        const float qv = e[(size_t)d * K_CODES + k];
        ee += ev * ev; qq += qv * qv; eq += ev * qv;
    }
#pragma unroll
    for (int m = 1; m < 32; m <<= 1) {
        ee += __shfl_xor(ee, m, 32);
        qq += __shfl_xor(qq, m, 32);
        eq += __shfl_xor(eq, m, 32);
    }
    if (lane == 0) {
        const float en  = fmaxf(sqrtf(ee), EPSF);
        const float qn  = fmaxf(sqrtf(qq), EPSF);
        const float sn2 = ee / (en * en) + 2.f * eq / (en * qn) + qq / (qn * qn);
        const float sn  = fmaxf(sqrtf(sn2), EPSF);
        const float rde = (ee / en + eq / qn) / sn;
        const float lam = qn / en;
        ab[2 * n + 0] = lam * (1.f - 2.f * rde / (en * sn));
        ab[2 * n + 1] = 2.f * lam * (en - rde / sn) / qn;
    }
}

// ---------------------------------------------------------------------------
// Kernel 6: out[b,d,hw] = alpha[n]*x[b,d,hw] + beta[n]*e[d,idx[n]]
// ---------------------------------------------------------------------------
__global__ __launch_bounds__(256)
void vq_apply(const float* __restrict__ x, const float* __restrict__ e,
              const int* __restrict__ idx, const float* __restrict__ ab,
              float* __restrict__ out) {
    const size_t t = (size_t)blockIdx.x * 256 + threadIdx.x;
    const int hw = (int)(t & (HW - 1));
    const int d  = (int)((t >> 12) & (D_DIM - 1));
    const int b  = (int)(t >> 20);
    const int n  = b * HW + hw;
    const int k  = idx[n];
    out[t] = ab[2 * n] * x[t] + ab[2 * n + 1] * e[(size_t)d * K_CODES + k];
}

// ---------------------------------------------------------------------------
extern "C" void kernel_launch(void* const* d_in, const int* in_sizes, int n_in,
                              void* d_out, int out_size, void* d_ws, size_t ws_size,
                              hipStream_t stream) {
    const float* x = (const float*)d_in[0];       // [8,256,64,64]
    const float* e = (const float*)d_in[1];       // [256,8192]
    float* outf    = (float*)d_out;               // [8,256,64,64] then indices
    int*   idx     = (int*)(outf + (size_t)B_DIM * D_DIM * HW);

    char* ws = (char*)d_ws;
    float*  Af = (float*)(ws);                               // 32 MiB
    bf16_t* Ab = (bf16_t*)(ws + 33554432);                   // 16 MiB
    bf16_t* Bb = (bf16_t*)(ws + 50331648);                   //  4 MiB
    float*  cn = (float*)(ws + 54525952);                    // 32 KiB
    float*  ab = (float*)(ws + 54558720);                    // 256 KiB

    dim3 tb(32, 8);
    vq_transpose_x <<<dim3(HW / 32, D_DIM / 32, B_DIM), tb, 0, stream>>>(x, Af, Ab);
    vq_transpose_cb<<<dim3(K_CODES / 32, D_DIM / 32),    tb, 0, stream>>>(e, Bb);
    vq_cnorm       <<<K_CODES / 256, 256, 0, stream>>>(e, cn);
    vq_argmin_wmma <<<N_ROWS / 256, 256, 0, stream>>>(Ab, Bb, cn, idx);
    vq_rowstats    <<<N_ROWS / 8,   256, 0, stream>>>(Af, e, idx, ab);
    vq_apply       <<<(B_DIM * D_DIM * HW) / 256, 256, 0, stream>>>(x, e, idx, ab, outf);
}
